// BiLinearAttention_74062416052581
// MI455X (gfx1250) — matrix operand order repaired and checked by reading, not compiled, not run on hardware
//
#include <hip/hip_runtime.h>
#include <hip/hip_bf16.h>
#include <math.h>

namespace {

constexpr int kN = 8;
constexpr int kL = 2048;
constexpr int kS = 2048;
constexpr int kD = 1024;

typedef __bf16 bf16;
typedef __attribute__((ext_vector_type(16))) __bf16 bf16x16;
typedef __attribute__((ext_vector_type(8)))  __bf16 bf16x8;
typedef __attribute__((ext_vector_type(4)))  __bf16 bf16x4;
typedef __attribute__((ext_vector_type(8)))  float  f32x8;

constexpr int kQStr = 1024 + 8;  // padded LDS row stride (bf16) for A staging
constexpr int kPStr = kS + 8;    // padded LDS row stride (bf16) for P buffer

__device__ __forceinline__ f32x8 zero_f32x8() {
  f32x8 z;
#pragma unroll
  for (int i = 0; i < 8; ++i) z[i] = 0.0f;
  return z;
}

__device__ __forceinline__ bf16x16 combine16(bf16x8 lo, bf16x8 hi) {
  return __builtin_shufflevector(lo, hi, 0, 1, 2, 3, 4, 5, 6, 7,
                                 8, 9, 10, 11, 12, 13, 14, 15);
}

// A-matrix fragment (16x32 bf16, row-major, K contiguous). ISA 7.12.2:
// lanes 0-15: row m, K {0..7, 16..23}; lanes 16-31: row m, K {8..15, 24..31}.
__device__ __forceinline__ bf16x16 frag_a(const bf16* rowp, int lane) {
  const bf16* p = rowp + ((lane & 16) ? 8 : 0);
  bf16x8 lo = *(const bf16x8*)(p);
  bf16x8 hi = *(const bf16x8*)(p + 16);
  return combine16(lo, hi);
}

// B-matrix fragment fed from B^T rows (16 rows x 32 K, K contiguous). ISA:
// lanes 0-15 hold column n=m, K {0..7, 16..23}; lanes 16-31 hold K {8..15, 24..31}.
__device__ __forceinline__ bf16x16 frag_b(const bf16* rowp, int lane) {
  const bf16* p = rowp + ((lane & 16) ? 8 : 0);
  bf16x8 lo = *(const bf16x8*)(p);
  bf16x8 hi = *(const bf16x8*)(p + 16);
  return combine16(lo, hi);
}

__device__ __forceinline__ f32x8 wmma_bf16(bf16x16 a, bf16x16 b, f32x8 c) {
  return __builtin_amdgcn_wmma_f32_16x16x32_bf16(
      false, a, false, b, (short)0, c, false, false);
}

// Cooperative stage of a 16x1024 bf16 A-tile (row-major) into LDS with
// padded row stride kQStr. 256 threads, coalesced b128 reads -> ds writes.
__device__ __forceinline__ void stage_a_tile(const bf16* __restrict__ src,
                                             bf16* dst) {
  int tid = threadIdx.x;
#pragma unroll
  for (int c = 0; c < 8; ++c) {
    int idx  = tid + c * 256;     // chunk id, 16 rows * 128 chunks
    int row  = idx >> 7;
    int col8 = idx & 127;
    *(bf16x8*)(dst + row * kQStr + col8 * 8) =
        *(const bf16x8*)(src + row * kD + col8 * 8);
  }
}

// ---------------- elementwise f32 -> bf16 ----------------
__global__ void cvt_kernel(const float* __restrict__ in,
                           bf16* __restrict__ out, int n4) {
  int i = blockIdx.x * blockDim.x + threadIdx.x;
  if (i < n4) {
    float4 v = ((const float4*)in)[i];
    bf16x4 o;
    o.x = (bf16)v.x; o.y = (bf16)v.y; o.z = (bf16)v.z; o.w = (bf16)v.w;
    ((bf16x4*)out)[i] = o;
  }
}

// ------------- value [N,S,D] f32 -> valueT [N,D,S] bf16 -------------
__global__ void transpose_cvt_kernel(const float* __restrict__ V,
                                     bf16* __restrict__ VT) {
  __shared__ bf16 tile[32][33];
  int n  = blockIdx.z;
  int s0 = blockIdx.x * 32;
  int d0 = blockIdx.y * 32;
  int tx = threadIdx.x & 31;
  int ty = threadIdx.x >> 5;  // 0..7
  const float* Vn = V + (size_t)n * kS * kD;
  bf16* VTn = VT + (size_t)n * kD * kS;
#pragma unroll
  for (int i = 0; i < 4; ++i) {
    int s = s0 + ty + 8 * i;
    tile[ty + 8 * i][tx] = (bf16)Vn[(size_t)s * kD + d0 + tx];
  }
  __syncthreads();
#pragma unroll
  for (int i = 0; i < 4; ++i) {
    int d = d0 + ty + 8 * i;
    VTn[(size_t)d * kS + s0 + tx] = tile[tx][ty + 8 * i];
  }
}

// ------------- q_proj = query @ W^T + b (bf16 WMMA, f32 acc) -------------
// One block = 16 rows of (N*L); 8 waves split D into 128-col strips.
__global__ __launch_bounds__(256) void proj_kernel(
    const bf16* __restrict__ Qb, const bf16* __restrict__ Wb,
    const float* __restrict__ bias, bf16* __restrict__ QP) {
  __shared__ bf16 QA[16 * kQStr];  // ~33 KB

  int row0 = blockIdx.x * 16;
  int wave = threadIdx.x >> 5;
  int lane = threadIdx.x & 31;
  int m    = lane & 15;
  int hi8  = (lane & 16) ? 8 : 0;
  int d0   = wave * 128;

  stage_a_tile(Qb + (size_t)row0 * kD, QA);
  __syncthreads();

  f32x8 acc[8];
#pragma unroll
  for (int t = 0; t < 8; ++t) acc[t] = zero_f32x8();

#pragma unroll 1
  for (int k0 = 0; k0 < kD; k0 += 32) {
    bf16x16 a = frag_a(&QA[m * kQStr + k0], lane);
    bf16x16 b[8];
#pragma unroll
    for (int t = 0; t < 8; ++t)
      b[t] = frag_b(Wb + (size_t)(d0 + 16 * t + m) * kD + k0, lane);
#pragma unroll
    for (int t = 0; t < 8; ++t) acc[t] = wmma_bf16(a, b[t], acc[t]);
  }
  // C/D layout: lane m = col, VGPR r = row r (lanes<16) / row 8+r (lanes>=16)
#pragma unroll
  for (int t = 0; t < 8; ++t) {
    int col  = d0 + 16 * t + m;
    float bv = bias[col];
#pragma unroll
    for (int r = 0; r < 8; ++r)
      QP[(size_t)(row0 + hi8 + r) * kD + col] = (bf16)(acc[t][r] + bv);
  }
}

// ------------- fused score GEMM + softmax + PV GEMM -------------
__global__ __launch_bounds__(256) void attn_kernel(
    const bf16* __restrict__ QP, const bf16* __restrict__ Kb,
    const bf16* __restrict__ VT, float* __restrict__ Out) {
  extern __shared__ char smem[];
  // QA (16*kQStr bf16 = ~33 KB) aliases the front of Pbuf (~66 KB): QA is
  // dead before Pbuf is written, with two __syncthreads in between.
  bf16*  QA   = (bf16*)smem;
  bf16*  Pbuf = (bf16*)smem;
  float* redA = (float*)(smem + (size_t)16 * kPStr * 2);  // 8 waves * 16 rows
  float* redB = redA + 8 * 16;

  int n    = blockIdx.y;
  int l0   = blockIdx.x * 16;
  int wave = threadIdx.x >> 5;
  int lane = threadIdx.x & 31;
  int m    = lane & 15;
  int hi8  = (lane & 16) ? 8 : 0;

  const bf16* QPn = QP + ((size_t)n * kL + l0) * kD;
  const bf16* Kn  = Kb + (size_t)n * kS * kD;
  const bf16* VTn = VT + (size_t)n * kD * kS;

  stage_a_tile(QPn, QA);
  __syncthreads();

  // ---- Phase A: scores[16 L x 256 S] per wave ----
  int sbase = wave * 256;
  f32x8 acc[16];
#pragma unroll
  for (int t = 0; t < 16; ++t) acc[t] = zero_f32x8();

#pragma unroll 1
  for (int k0 = 0; k0 < kD; k0 += 32) {
    bf16x16 a = frag_a(&QA[m * kQStr + k0], lane);
    bf16x16 b[16];
#pragma unroll
    for (int t = 0; t < 16; ++t)
      b[t] = frag_b(Kn + (size_t)(sbase + 16 * t + m) * kD + k0, lane);
#pragma unroll
    for (int t = 0; t < 16; ++t) acc[t] = wmma_bf16(a, b[t], acc[t]);
  }

  // ---- Phase B: exact softmax over full S ----
  float rmax[8], rsum[8];
#pragma unroll
  for (int r = 0; r < 8; ++r) {
    float v = acc[0][r];
#pragma unroll
    for (int t = 1; t < 16; ++t) v = fmaxf(v, acc[t][r]);
    v = fmaxf(v, __shfl_xor(v, 1));
    v = fmaxf(v, __shfl_xor(v, 2));
    v = fmaxf(v, __shfl_xor(v, 4));
    v = fmaxf(v, __shfl_xor(v, 8));  // reduce within 16-lane half => per row
    rmax[r] = v;
  }
  if (m == 0) {
#pragma unroll
    for (int r = 0; r < 8; ++r) redA[wave * 16 + hi8 + r] = rmax[r];
  }
  __syncthreads();
#pragma unroll
  for (int r = 0; r < 8; ++r) {
    float v = redA[hi8 + r];
#pragma unroll
    for (int w = 1; w < 8; ++w) v = fmaxf(v, redA[w * 16 + hi8 + r]);
    rmax[r] = v;
  }
  const float kLog2e = 1.4426950408889634f;
#pragma unroll
  for (int r = 0; r < 8; ++r) {
    float s = 0.0f;
#pragma unroll
    for (int t = 0; t < 16; ++t) {
      float p = exp2f((acc[t][r] - rmax[r]) * kLog2e);
      acc[t][r] = p;
      s += p;
    }
    s += __shfl_xor(s, 1);
    s += __shfl_xor(s, 2);
    s += __shfl_xor(s, 4);
    s += __shfl_xor(s, 8);
    rsum[r] = s;
  }
  if (m == 0) {
#pragma unroll
    for (int r = 0; r < 8; ++r) redB[wave * 16 + hi8 + r] = rsum[r];
  }
  __syncthreads();
#pragma unroll
  for (int r = 0; r < 8; ++r) {
    float s = 0.0f;
#pragma unroll
    for (int w = 0; w < 8; ++w) s += redB[w * 16 + hi8 + r];
    rsum[r] = 1.0f / s;
  }
  // write normalized P (bf16) into LDS
#pragma unroll
  for (int t = 0; t < 16; ++t) {
#pragma unroll
    for (int r = 0; r < 8; ++r)
      Pbuf[(hi8 + r) * kPStr + sbase + 16 * t + m] = (bf16)(acc[t][r] * rsum[r]);
  }
  __syncthreads();

  // ---- Phase C: out[16 L x 128 D] per wave = P @ value ----
  int dw0 = wave * 128;
  f32x8 oacc[8];
#pragma unroll
  for (int t = 0; t < 8; ++t) oacc[t] = zero_f32x8();

#pragma unroll 1
  for (int k0 = 0; k0 < kS; k0 += 32) {
    bf16x16 a = frag_a(&Pbuf[m * kPStr + k0], lane);  // ds_read_b128
    bf16x16 b[8];
#pragma unroll
    for (int t = 0; t < 8; ++t)
      b[t] = frag_b(VTn + (size_t)(dw0 + 16 * t + m) * kS + k0, lane);
#pragma unroll
    for (int t = 0; t < 8; ++t) oacc[t] = wmma_bf16(a, b[t], oacc[t]);
  }

  float* On = Out + ((size_t)n * kL + l0) * kD;
#pragma unroll
  for (int t = 0; t < 8; ++t) {
#pragma unroll
    for (int r = 0; r < 8; ++r)
      On[(size_t)(hi8 + r) * kD + dw0 + 16 * t + m] = oacc[t][r];
  }
}

}  // namespace

extern "C" void kernel_launch(void* const* d_in, const int* in_sizes, int n_in,
                              void* d_out, int out_size, void* d_ws,
                              size_t ws_size, hipStream_t stream) {
  (void)in_sizes; (void)n_in; (void)out_size; (void)ws_size;
  const float* key   = (const float*)d_in[0];
  const float* query = (const float*)d_in[1];
  const float* value = (const float*)d_in[2];
  const float* W     = (const float*)d_in[3];
  const float* bias  = (const float*)d_in[4];
  float* out = (float*)d_out;

  const size_t nld = (size_t)kN * kL * kD;
  const size_t nsd = (size_t)kN * kS * kD;
  char* ws = (char*)d_ws;
  size_t off = 0;
  bf16* Qb = (bf16*)(ws + off); off += nld * 2;            // query bf16
  bf16* Kb = (bf16*)(ws + off); off += nsd * 2;            // key bf16
  bf16* VT = (bf16*)(ws + off); off += nsd * 2;            // value^T bf16
  bf16* Wb = (bf16*)(ws + off); off += (size_t)kD * kD * 2;
  bf16* QP = (bf16*)(ws + off); off += nld * 2;            // q_proj bf16
  // total ~130 MB scratch

  int nq4 = (int)(nld / 4);
  int nk4 = (int)(nsd / 4);
  int nw4 = kD * kD / 4;
  cvt_kernel<<<(nq4 + 255) / 256, 256, 0, stream>>>(query, Qb, nq4);
  cvt_kernel<<<(nk4 + 255) / 256, 256, 0, stream>>>(key, Kb, nk4);
  cvt_kernel<<<(nw4 + 255) / 256, 256, 0, stream>>>(W, Wb, nw4);
  transpose_cvt_kernel<<<dim3(kS / 32, kD / 32, kN), 256, 0, stream>>>(value, VT);

  proj_kernel<<<kN * kL / 16, 256, 0, stream>>>(Qb, Wb, bias, QP);

  int smem = 16 * kPStr * 2 + 2 * 8 * 16 * (int)sizeof(float);  // ~66.8 KB
  hipFuncSetAttribute((const void*)attn_kernel,
                      hipFuncAttributeMaxDynamicSharedMemorySize, smem);
  attn_kernel<<<dim3(kL / 16, kN), 256, smem, stream>>>(QP, Kb, VT, out);
}
